// SelfAttention_38697655337053
// MI455X (gfx1250) — matrix operand order repaired and checked by reading, not compiled, not run on hardware
//
#include <hip/hip_runtime.h>

// Self-attention forward for B=8, N=2048, D=F=768 on gfx1250 (MI455X).
// All GEMMs via v_wmma_f32_16x16x32_f16 (f32 accumulate), flash-attention
// second stage (scores never touch HBM). Workspace layout (f16 elements):
//   Xh[B*N*D] | Qh[B*N*F] | Kh[B*N*F] | Vt[B*D*N] | WqT[F*D] | WkT[F*D] | WvT[D*D]
// Total ~104 MB of d_ws.

typedef __attribute__((ext_vector_type(16))) _Float16 v16h;
typedef __attribute__((ext_vector_type(8)))  float    v8f;
typedef int b128v __attribute__((vector_size(16)));   // matches async-LDS builtin

constexpr int B_ = 8, N_ = 2048, D_ = 768, F_ = 768;

#define GLOBAL_AS __attribute__((address_space(1)))
#define LDS_AS    __attribute__((address_space(3)))

#if __has_builtin(__builtin_amdgcn_global_load_async_to_lds_b128) && \
    __has_builtin(__builtin_amdgcn_s_wait_asynccnt)
#define USE_ASYNC_LDS 1
#else
#define USE_ASYNC_LDS 0
#endif

union V16U { v16h v; uint4 q[2]; };

__device__ __forceinline__ v8f wmma16(v16h a, v16h b, v8f c) {
  // 8 args: (neg_a, A, neg_b, B, c_mod, C, reuse_a, reuse_b)
  return __builtin_amdgcn_wmma_f32_16x16x32_f16(false, a, false, b, (short)0, c,
                                                false, false);
}

// B-operand load: 16 contiguous halves (lane half selects low/high K half).
__device__ __forceinline__ v16h ld16(const _Float16* p) {
  V16U u;
  u.q[0] = *(const uint4*)p;
  u.q[1] = *(const uint4*)(p + 8);
  return u.v;
}
// A-operand load: two 8-half chunks at p and p+16 halves (ISA A layout:
// lane<16 holds K {k0..k0+7, k0+16..k0+23}; caller adds half*8 to base).
__device__ __forceinline__ v16h ld16_split(const _Float16* p) {
  V16U u;
  u.q[0] = *(const uint4*)p;
  u.q[1] = *(const uint4*)(p + 16);
  return u.v;
}

// ---------------- Kernel 1: X fp32 -> f16 ----------------
__global__ void cvt_x(const float* __restrict__ x, _Float16* __restrict__ xh) {
  size_t i = ((size_t)blockIdx.x * blockDim.x + threadIdx.x) * 4;
  float4 v = *(const float4*)(x + i);
  union { _Float16 h[4]; uint2 u; } pk;
  pk.h[0] = (_Float16)v.x;
  pk.h[1] = (_Float16)v.y;
  pk.h[2] = (_Float16)v.z;
  pk.h[3] = (_Float16)v.w;
  *(uint2*)&xh[i] = pk.u;          // one 8B store instead of 4x b16
}

// ---------------- Kernel 2: W -> W^T f16 ----------------
__global__ void cvt_w(const float* __restrict__ wq, const float* __restrict__ wk,
                      const float* __restrict__ wv, _Float16* __restrict__ wqt,
                      _Float16* __restrict__ wkt, _Float16* __restrict__ wvt) {
  int i = blockIdx.x * blockDim.x + threadIdx.x;   // [0, D*F)
  int j = i / D_;                                  // output row = W column
  int k = i % D_;
  int m = blockIdx.y;
  const float* src = (m == 0) ? wq : (m == 1) ? wk : wv;
  _Float16* dst    = (m == 0) ? wqt : (m == 1) ? wkt : wvt;
  dst[(size_t)j * D_ + k] = (_Float16)src[(size_t)k * F_ + j];
}

// ---------------- Kernel 3: Q/K/V projections ----------------
// grid.x = B*N/16 row tiles, grid.y = 18; 8 waves/block, wave -> (matrix, coltile).
// Vt written transposed: Vt[b][d][n] via packed b128 stores (C-layout gives each
// lane 8 consecutive n for a fixed d).
__global__ __launch_bounds__(256) void proj_qkv(
    const _Float16* __restrict__ xh, const _Float16* __restrict__ wqt,
    const _Float16* __restrict__ wkt, const _Float16* __restrict__ wvt,
    _Float16* __restrict__ qh, _Float16* __restrict__ kh,
    _Float16* __restrict__ vt) {
  constexpr int LDA = 776;                 // 768 + 8 halves: bank-conflict pad
  __shared__ _Float16 xs[16 * LDA];

  const int row0 = blockIdx.x * 16;        // flat row in [0, B*N)

  // Stage 16x768 f16 X slab (6 x 16B per thread).
#if USE_ASYNC_LDS
  for (int c = threadIdx.x; c < 16 * 96; c += 256) {
    int r = c / 96, off = (c % 96) * 8;
    __builtin_amdgcn_global_load_async_to_lds_b128(
        (GLOBAL_AS b128v*)&xh[(size_t)(row0 + r) * D_ + off],
        (LDS_AS b128v*)&xs[r * LDA + off], 0, 0);
  }
  __builtin_amdgcn_s_wait_asynccnt(0);
  __syncthreads();
#else
  for (int c = threadIdx.x; c < 16 * 96; c += 256) {
    int r = c / 96, off = (c % 96) * 8;
    *(uint4*)&xs[r * LDA + off] = *(const uint4*)&xh[(size_t)(row0 + r) * D_ + off];
  }
  __syncthreads();
#endif

  const int wave = threadIdx.x >> 5, lane = threadIdx.x & 31;
  const int half = lane >> 4, l16 = lane & 15;
  const int combo = blockIdx.y * 8 + wave;         // [0, 144)
  const int mat = combo / 48, ct = combo % 48;
  const _Float16* wt = (mat == 0) ? wqt : (mat == 1) ? wkt : wvt;
  const _Float16* wrow = wt + (size_t)(ct * 16 + l16) * D_;

  v8f acc = {};
#pragma unroll
  for (int kk = 0; kk < D_ / 32; ++kk) {
    v16h a = ld16(&xs[l16 * LDA + kk * 32 + half * 16]);
    v16h bm = ld16(&wrow[kk * 32 + half * 16]);
    acc = wmma16(a, bm, acc);
  }

  if (mat < 2) {
    _Float16* out = (mat == 0) ? qh : kh;
    const int col = ct * 16 + l16;
#pragma unroll
    for (int r = 0; r < 8; ++r) {
      int m = r + half * 8;
      out[(size_t)(row0 + m) * F_ + col] = (_Float16)acc[r];  // 32B/lane-group
    }
  } else {
    // C rows m = half*8 + r at fixed column d = ct*16+l16 are contiguous in n
    // for Vt[d][n]: pack 8 halves and store one b128 per lane.
    const int b = row0 / N_, n0 = row0 % N_;
    const int d = ct * 16 + l16;
    union { _Float16 h[8]; uint4 u; } pk;
#pragma unroll
    for (int r = 0; r < 8; ++r) pk.h[r] = (_Float16)acc[r];
    *(uint4*)&vt[((size_t)b * D_ + d) * N_ + n0 + half * 8] = pk.u;
  }
}

// ---------------- Kernel 4: flash attention ----------------
// Block: 16 queries of one batch, 4 waves; wave w owns F/D slice [w*192, w*192+192).
// Per 32-key iteration: each wave does 12 WMMAs of partial S over its F slice,
// LDS-reduce S, identical online softmax in all waves, wave0 emits P (f16,
// A-operand order) to LDS, each wave does 12 P*V WMMAs into its O slice with
// the rescaled O as the WMMA C operand. Zero redundant matrix work; each
// wave's K/V B-operand global reads are disjoint (L2 absorbs re-reads).
__global__ __launch_bounds__(128) void flash_attn(
    const _Float16* __restrict__ qh, const _Float16* __restrict__ kh,
    const _Float16* __restrict__ vt, float* __restrict__ out) {
  constexpr int LDSP = 20;   // dwords per lane in S-partial buffer (pad: 16->20)
  constexpr int LDP  = 40;   // halves per row of P buffer (pad: 32->40)
  __shared__ float    sp[4 * 32 * LDSP];     // 10.0 KB
  __shared__ _Float16 pbuf[16 * LDP];        // 1.25 KB

  const int qt = blockIdx.x, b = blockIdx.y;
  const int wave = threadIdx.x >> 5, lane = threadIdx.x & 31;
  const int half = lane >> 4, l16 = lane & 15;
  const int q0 = qt * 16;
  const int f0 = wave * 192;                 // this wave's F/D slice

  const _Float16* qb = qh + ((size_t)b * N_ + q0) * F_;
  const _Float16* kb = kh + (size_t)b * N_ * F_;
  const _Float16* vb = vt + (size_t)b * D_ * N_;

  // Q A-tiles for this wave's F slice (held in VGPRs for the whole key loop).
  v16h qa[6];
#pragma unroll
  for (int c = 0; c < 6; ++c)
    qa[c] = ld16(&qb[(size_t)l16 * F_ + f0 + c * 32 + half * 16]);

  v8f o[12];
#pragma unroll
  for (int t = 0; t < 12; ++t) o[t] = (v8f){};
  float rm[8], rl[8];
#pragma unroll
  for (int r = 0; r < 8; ++r) { rm[r] = -1e30f; rl[r] = 0.f; }

  float* mySp = sp + (wave * 32 + lane) * LDSP;

  for (int it = 0; it < N_ / 32; ++it) {
    const int key0 = it * 32;

    if (key0 + 32 < N_) {  // pull next K/V tiles toward L2/L0
      __builtin_prefetch(kb + (size_t)(key0 + 32 + l16) * F_ + f0, 0, 1);
      __builtin_prefetch(vb + (size_t)(f0 + l16) * N_ + key0 + 32, 0, 1);
    }

    // ---- partial S = Q * K^T over this wave's F slice, 2 key subtiles ----
    v8f s0 = {}, s1 = {};
#pragma unroll
    for (int c = 0; c < 6; ++c) {
      const int f = f0 + c * 32 + half * 16;
      v16h b0 = ld16(&kb[(size_t)(key0 + l16) * F_ + f]);
      v16h b1 = ld16(&kb[(size_t)(key0 + 16 + l16) * F_ + f]);
      s0 = wmma16(qa[c], b0, s0);
      s1 = wmma16(qa[c], b1, s1);
    }
#pragma unroll
    for (int r = 0; r < 8; r += 4) {
      *(float4*)&mySp[r]     = make_float4(s0[r], s0[r + 1], s0[r + 2], s0[r + 3]);
      *(float4*)&mySp[8 + r] = make_float4(s1[r], s1[r + 1], s1[r + 2], s1[r + 3]);
    }
    __syncthreads();

    // ---- reduce S partials over the 4 waves ----
    float sf[16];
#pragma unroll
    for (int i = 0; i < 16; ++i) sf[i] = 0.f;
#pragma unroll
    for (int w2 = 0; w2 < 4; ++w2) {
      const float* p = sp + (w2 * 32 + lane) * LDSP;
#pragma unroll
      for (int i = 0; i < 16; i += 4) {
        float4 v = *(const float4*)&p[i];
        sf[i] += v.x; sf[i + 1] += v.y; sf[i + 2] += v.z; sf[i + 3] += v.w;
      }
    }

    // ---- online softmax (identical in all waves). Row r+8*half lives in
    // VGPR r across a 16-lane half: shfl_xor {1,2,4,8} reduces per row. ----
    float nm[8], alpha[8];
#pragma unroll
    for (int r = 0; r < 8; ++r) {
      float mx = fmaxf(sf[r], sf[8 + r]);
#pragma unroll
      for (int d = 1; d < 16; d <<= 1) mx = fmaxf(mx, __shfl_xor(mx, d, 32));
      nm[r] = fmaxf(rm[r], mx);
      alpha[r] = __expf(rm[r] - nm[r]);
    }
#pragma unroll
    for (int r = 0; r < 8; ++r) {
      sf[r]     = __expf(sf[r] - nm[r]);
      sf[8 + r] = __expf(sf[8 + r] - nm[r]);
      float s = sf[r] + sf[8 + r];
#pragma unroll
      for (int d = 1; d < 16; d <<= 1) s += __shfl_xor(s, d, 32);
      rl[r] = rl[r] * alpha[r] + s;
      rm[r] = nm[r];
    }

    // ---- wave0 writes P (f16) in row-major for A-operand pickup ----
    if (wave == 0) {
#pragma unroll
      for (int r = 0; r < 8; ++r) {
        int m = r + half * 8;
        pbuf[m * LDP + l16]      = (_Float16)sf[r];
        pbuf[m * LDP + 16 + l16] = (_Float16)sf[8 + r];
      }
    }
    __syncthreads();

    // ---- O = P * V + alpha*O over this wave's 192-wide D slice ----
    v16h pa = ld16(&pbuf[l16 * LDP + half * 16]);
#pragma unroll
    for (int t = 0; t < 12; ++t) {
      v16h vv = ld16(&vb[(size_t)(f0 + t * 16 + l16) * N_ + key0 + half * 16]);
      v8f c;
#pragma unroll
      for (int r = 0; r < 8; ++r) c[r] = o[t][r] * alpha[r];
      o[t] = wmma16(pa, vv, c);
    }
    __syncthreads();
  }

  // ---- normalize and write fp32 output (64B segments per lane group) ----
  float* ob = out + ((size_t)b * N_ + q0) * D_;
  float inv[8];
#pragma unroll
  for (int r = 0; r < 8; ++r) inv[r] = 1.f / rl[r];
#pragma unroll
  for (int t = 0; t < 12; ++t) {
    const int d = f0 + t * 16 + l16;
#pragma unroll
    for (int r = 0; r < 8; ++r) {
      int m = r + half * 8;
      ob[(size_t)m * D_ + d] = o[t][r] * inv[r];
    }
  }
}

// ---------------- host launcher ----------------
extern "C" void kernel_launch(void* const* d_in, const int* in_sizes, int n_in,
                              void* d_out, int out_size, void* d_ws, size_t ws_size,
                              hipStream_t stream) {
  const float* x  = (const float*)d_in[0];
  const float* wq = (const float*)d_in[1];
  const float* wk = (const float*)d_in[2];
  const float* wv = (const float*)d_in[3];
  float* out = (float*)d_out;

  const size_t XN = (size_t)B_ * N_ * D_;   // 12,582,912
  _Float16* xh  = (_Float16*)d_ws;
  _Float16* qh  = xh + XN;
  _Float16* kh  = qh + XN;
  _Float16* vt  = kh + XN;
  _Float16* wqt = vt + XN;
  _Float16* wkt = wqt + (size_t)F_ * D_;
  _Float16* wvt = wkt + (size_t)F_ * D_;
  // total ws use: (4*XN + 3*F*D) * 2 bytes ~= 104 MB

  cvt_x<<<dim3((unsigned)(XN / 4 / 256)), 256, 0, stream>>>(x, xh);
  cvt_w<<<dim3(D_ * F_ / 256, 3), 256, 0, stream>>>(wq, wk, wv, wqt, wkt, wvt);
  proj_qkv<<<dim3(B_ * N_ / 16, 18), 256, 0, stream>>>(xh, wqt, wkt, wvt, qh, kh, vt);
  flash_attn<<<dim3(N_ / 16, B_), 128, 0, stream>>>(qh, kh, vt, out);
}
